// TinyMamba_44848048505421
// MI455X (gfx1250) — compile-verified, matrix operand order checked by reading, not run on hardware
//
#include <hip/hip_runtime.h>
#include <hip/hip_bf16.h>

// ---------------- constants ----------------
#define BATCH     32
#define SEQL      2048
#define BL        (BATCH * SEQL)        // 65536 tokens
#define IN_DIM    57
#define KPAD      64                    // IN_DIM padded for WMMA K
#define D_MODEL   256
#define D_INNER   512
#define XZ_COLS   1024                  // 2 * D_INNER
#define D_STATE   16
#define DT_RANK   16
#define XD_COLS   48                    // DT_RANK + 2*D_STATE
#define XD_PAD    64
#define D_CONV    4
#define OUTC      6
#define CHUNK     128                   // scan chunk length
#define NCHUNK    (SEQL / CHUNK)        // 16 chunks per batch

typedef __attribute__((ext_vector_type(16))) _Float16 v16h;
typedef __attribute__((ext_vector_type(8)))  _Float16 v8h;
typedef __attribute__((ext_vector_type(8)))  float    v8f;

// Packed B-fragment layouts (one 16-half chunk per lane, contiguous):
//   WfT: [n_tile(64)][k_step(2)][lane(32)][elem(16)]   halves
//   WxT: [n_tile(4)][k_step(16)][lane(32)][elem(16)]   halves
#define WFT_ELEMS (64 * 2 * 32 * 16)    // 65536
#define WXT_ELEMS (4 * 16 * 32 * 16)    // 32768

// Map WMMA 16-bit B-fragment (lane, elem) -> K index within a 32-wide k-step.
__device__ __forceinline__ int frag_k(int lane, int i) {
  int krb = (lane < 16) ? 0 : 8;
  return krb + ((i < 8) ? i : (16 + i - 8));
}

// ---------------------------------------------------------------------------
// K0: build WfT = swizzle(W_in@W_inproj, f16, K padded to 64),
//     bfus = b_in@W_inproj, WxT = swizzle(pad(W_xproj)), Woc = W_out@W_cls.
// ---------------------------------------------------------------------------
__global__ void k_prep(const float* __restrict__ W_in, const float* __restrict__ b_in,
                       const float* __restrict__ W_inproj, const float* __restrict__ W_xproj,
                       const float* __restrict__ W_out, const float* __restrict__ W_cls,
                       _Float16* __restrict__ WfT, float* __restrict__ bfus,
                       _Float16* __restrict__ WxT, float* __restrict__ Woc) {
  int tid = blockIdx.x * blockDim.x + threadIdx.x;
  if (tid < WFT_ELEMS) {                            // fused input weight, swizzled
    int i  = tid & 15;
    int l  = (tid >> 4) & 31;
    int ks = (tid >> 9) & 1;
    int nt = tid >> 10;                             // 0..63
    int col = nt * 16 + (l & 15);
    int k   = ks * 32 + frag_k(l, i);
    float acc = 0.f;
    if (k < IN_DIM)
      for (int m = 0; m < D_MODEL; ++m)
        acc += W_in[k * D_MODEL + m] * W_inproj[m * XZ_COLS + col];
    WfT[tid] = (_Float16)acc;
    return;
  }
  tid -= WFT_ELEMS;
  if (tid < XZ_COLS) {                              // fused bias
    float acc = 0.f;
    for (int m = 0; m < D_MODEL; ++m) acc += b_in[m] * W_inproj[m * XZ_COLS + tid];
    bfus[tid] = acc;
    return;
  }
  tid -= XZ_COLS;
  if (tid < WXT_ELEMS) {                            // x-proj weight, swizzled + padded
    int i  = tid & 15;
    int l  = (tid >> 4) & 31;
    int ks = (tid >> 9) & 15;
    int nt = tid >> 13;                             // 0..3
    int col = nt * 16 + (l & 15);
    int k   = ks * 32 + frag_k(l, i);
    WxT[tid] = (_Float16)((col < XD_COLS) ? W_xproj[k * XD_COLS + col] : 0.f);
    return;
  }
  tid -= WXT_ELEMS;
  if (tid < D_INNER * OUTC) {                       // fused output weight
    int d = tid / OUTC, o = tid % OUTC;
    float acc = 0.f;
    for (int m = 0; m < D_MODEL; ++m) acc += W_out[d * D_MODEL + m] * W_cls[m * OUTC + o];
    Woc[tid] = acc;
  }
}

// ---------------------------------------------------------------------------
// K1: convert x (f32, 57/row) -> xh (f16, padded to 64/row)
// ---------------------------------------------------------------------------
__global__ void k_padx(const float* __restrict__ x, _Float16* __restrict__ xh) {
  size_t tid = (size_t)blockIdx.x * blockDim.x + threadIdx.x;   // BL*KPAD
  size_t row = tid >> 6;
  int k = (int)(tid & 63);
  xh[tid] = (_Float16)((k < IN_DIM) ? x[row * IN_DIM + k] : 0.f);
}

// ---------------------------------------------------------------------------
// K2: xz = xh @ Wf + bfus     (65536 x 64) @ (64 x 1024), f16 WMMA, f16 out.
// Block = 8 waves; each wave owns one 16-col n-tile (B frags = 2 vector loads,
// cached in VGPRs) and iterates 4 m-tiles. Grid: (1024/128, 65536/64).
// ---------------------------------------------------------------------------
__global__ __launch_bounds__(256) void k_gemm_xz(const _Float16* __restrict__ xh,
                                                 const _Float16* __restrict__ WfT,
                                                 const float* __restrict__ bfus,
                                                 _Float16* __restrict__ xzh) {
  const int lane = threadIdx.x & 31;
  const int wave = threadIdx.x >> 5;
  const int nt = blockIdx.x * 8 + wave;             // global n-tile, 0..63
  const int m_base = blockIdx.y * 64;
  const int col = nt * 16 + (lane & 15);
  const int krb = (lane < 16) ? 0 : 8;              // ISA 16-bit A lane K split

  // B fragments: one 32-byte contiguous load per k-step (coalesced per wave)
  const v16h b0 = *(const v16h*)(WfT + ((size_t)(nt * 2 + 0) * 32 + lane) * 16);
  const v16h b1 = *(const v16h*)(WfT + ((size_t)(nt * 2 + 1) * 32 + lane) * 16);
  const float bias = bfus[col];

#pragma unroll
  for (int mt = 0; mt < 4; ++mt) {
    const int m0 = m_base + mt * 16;
    const int row = m0 + (lane & 15);
    const _Float16* ap = xh + (size_t)row * KPAD + krb;
    v8h a_lo0 = *(const v8h*)(ap + 0);
    v8h a_hi0 = *(const v8h*)(ap + 16);
    v8h a_lo1 = *(const v8h*)(ap + 32);
    v8h a_hi1 = *(const v8h*)(ap + 48);
    v16h a0, a1;
#pragma unroll
    for (int t = 0; t < 8; ++t) {
      a0[t] = a_lo0[t]; a0[t + 8] = a_hi0[t];
      a1[t] = a_lo1[t]; a1[t + 8] = a_hi1[t];
    }
    v8f c = {};
    c = __builtin_amdgcn_wmma_f32_16x16x32_f16(false, a0, false, b0, (short)0, c, false, false);
    c = __builtin_amdgcn_wmma_f32_16x16x32_f16(false, a1, false, b1, (short)0, c, false, false);
    const int rbase = m0 + ((lane < 16) ? 0 : 8);
#pragma unroll
    for (int r = 0; r < 8; ++r)
      xzh[(size_t)(rbase + r) * XZ_COLS + col] = (_Float16)(c[r] + bias);
  }
}

// ---------------------------------------------------------------------------
// K3: depthwise causal conv (k=4) + SiLU over xin = xz[:, :512] -> xc (f16)
// ---------------------------------------------------------------------------
__global__ void k_conv(const _Float16* __restrict__ xzh, const float* __restrict__ conv_w,
                       const float* __restrict__ conv_b, _Float16* __restrict__ xch) {
  size_t tid = (size_t)blockIdx.x * blockDim.x + threadIdx.x;   // BL * D_INNER
  int d = (int)(tid & (D_INNER - 1));
  size_t row = tid >> 9;
  int l = (int)(row & (SEQL - 1));
  size_t base = row * XZ_COLS + d;                  // xin column d of token `row`
  float acc = conv_b[d];
#pragma unroll
  for (int j = 0; j < D_CONV; ++j) {
    int ll = l - (D_CONV - 1) + j;
    if (ll >= 0) acc += conv_w[d * D_CONV + j] * (float)xzh[base - (size_t)(D_CONV - 1 - j) * XZ_COLS];
  }
  xch[tid] = (_Float16)(acc / (1.f + __expf(-acc)));   // SiLU
}

// ---------------------------------------------------------------------------
// K4: x_dbl = xc @ Wx   (65536 x 512) @ (512 x 64), f16 WMMA, f32 out.
// Block = 8 waves: waves split 4 n-tiles x 2 m-tiles. Grid: 65536/32.
// ---------------------------------------------------------------------------
__global__ __launch_bounds__(256) void k_gemm_xdbl(const _Float16* __restrict__ xch,
                                                   const _Float16* __restrict__ WxT,
                                                   float* __restrict__ xdbl) {
  const int lane = threadIdx.x & 31;
  const int wave = threadIdx.x >> 5;
  const int ntile = wave & 3;
  const int msub  = wave >> 2;
  const int m0 = blockIdx.x * 32 + msub * 16;
  const int col = ntile * 16 + (lane & 15);
  const int krb = (lane < 16) ? 0 : 8;
  const int row = m0 + (lane & 15);

  v8f c = {};
#pragma unroll
  for (int ks = 0; ks < D_INNER / 32; ++ks) {
    const int koff = ks * 32;
    const v16h b = *(const v16h*)(WxT + ((size_t)(ntile * 16 + ks) * 32 + lane) * 16);
    const _Float16* ap = xch + (size_t)row * D_INNER + koff + krb;
    v8h alo = *(const v8h*)(ap);
    v8h ahi = *(const v8h*)(ap + 16);
    v16h a;
#pragma unroll
    for (int t = 0; t < 8; ++t) { a[t] = alo[t]; a[t + 8] = ahi[t]; }
    c = __builtin_amdgcn_wmma_f32_16x16x32_f16(false, a, false, b, (short)0, c, false, false);
  }
  const int rbase = m0 + ((lane < 16) ? 0 : 8);
#pragma unroll
  for (int r = 0; r < 8; ++r)
    xdbl[(size_t)(rbase + r) * XD_PAD + col] = c[r];
}

// ---------------------------------------------------------------------------
// K5a: chunked selective scan, pass A.  Grid (2, NCHUNK, BATCH), 256 thr.
// Per chunk (h0 = 0) each lane (channel d) computes:
//   h_end[n]  : partial end-state
//   P[n]      : decay product  prod_t exp(dt_t * A[n])
//   S[n]      : gated correction  sum_t silu(z_t) * C_t[n] * P_t[n]
//   accp      : partial gated output sum_t silu(z_t)*(y_t + D*u_t)
// True contribution given incoming h0:  accp + sum_n h0[n]*S[n];
// outgoing state: P o h0 + h_end.
// ---------------------------------------------------------------------------
__global__ __launch_bounds__(256) void k_scan_chunk(const _Float16* __restrict__ xzh,
                                                    const _Float16* __restrict__ xch,
                                                    const float* __restrict__ xdbl,
                                                    const float* __restrict__ W_dt,
                                                    const float* __restrict__ dt_bias,
                                                    const float* __restrict__ A_log,
                                                    const float* __restrict__ Dp,
                                                    float* __restrict__ CS_H,
                                                    float* __restrict__ CS_P,
                                                    float* __restrict__ CS_S,
                                                    float* __restrict__ CS_ACC) {
  const int d  = blockIdx.x * blockDim.x + threadIdx.x;   // 0..511
  const int c  = blockIdx.y;
  const int b  = blockIdx.z;
  const int bc = b * NCHUNK + c;
  __shared__ float s_in[XD_COLS];                   // dt_raw[16] | B[16] | C[16]

  float wdt[DT_RANK], Ad[D_STATE], h[D_STATE], P[D_STATE], Sg[D_STATE];
#pragma unroll
  for (int r = 0; r < DT_RANK; ++r) wdt[r] = W_dt[r * D_INNER + d];
#pragma unroll
  for (int n = 0; n < D_STATE; ++n) {
    Ad[n] = -__expf(A_log[d * D_STATE + n]);
    h[n] = 0.f; P[n] = 1.f; Sg[n] = 0.f;
  }
  const float bias = dt_bias[d];
  const float Dd = Dp[d];
  float accp = 0.f;

  for (int t = 0; t < CHUNK; ++t) {
    const size_t row = (size_t)b * SEQL + c * CHUNK + t;
    if (threadIdx.x < XD_COLS && blockIdx.x == 0) {}  // (all blocks load; cheap)
    if (threadIdx.x < XD_COLS) s_in[threadIdx.x] = xdbl[row * XD_PAD + threadIdx.x];
    if (t + 1 < CHUNK)                              // prefetch next timestep's u
      __builtin_prefetch(&xch[(row + 1) * D_INNER + d], 0, 0);
    __syncthreads();

    float dtr = bias;
#pragma unroll
    for (int r = 0; r < DT_RANK; ++r) dtr += s_in[r] * wdt[r];
    const float dt = (dtr > 20.f) ? dtr : __logf(1.f + __expf(dtr));  // softplus

    const float u  = (float)xch[row * D_INNER + d];
    const float zf = (float)xzh[row * XZ_COLS + D_INNER + d];
    const float sz = zf / (1.f + __expf(-zf));      // silu(z)

    float y = 0.f;
#pragma unroll
    for (int n = 0; n < D_STATE; ++n) {
      const float a  = __expf(dt * Ad[n]);
      h[n] = a * h[n] + dt * s_in[DT_RANK + n] * u;
      P[n] *= a;
      y += h[n] * s_in[DT_RANK + D_STATE + n];
      Sg[n] += sz * s_in[DT_RANK + D_STATE + n] * P[n];
    }
    accp += (y + Dd * u) * sz;
    __syncthreads();
  }
#pragma unroll
  for (int n = 0; n < D_STATE; ++n) {
    const size_t idx = ((size_t)bc * D_STATE + n) * D_INNER + d;
    CS_H[idx] = h[n];
    CS_P[idx] = P[n];
    CS_S[idx] = Sg[n];
  }
  CS_ACC[(size_t)bc * D_INNER + d] = accp;
}

// ---------------------------------------------------------------------------
// K5b: scan fix-up: serial over NCHUNK chunks, stitched per (b, d) lane.
// ---------------------------------------------------------------------------
__global__ __launch_bounds__(512) void k_scan_fix(const float* __restrict__ CS_H,
                                                  const float* __restrict__ CS_P,
                                                  const float* __restrict__ CS_S,
                                                  const float* __restrict__ CS_ACC,
                                                  float* __restrict__ ybar) {
  const int b = blockIdx.x;
  const int d = threadIdx.x;
  float h0[D_STATE];
#pragma unroll
  for (int n = 0; n < D_STATE; ++n) h0[n] = 0.f;
  float acc = 0.f;
  for (int c = 0; c < NCHUNK; ++c) {
    const int bc = b * NCHUNK + c;
    acc += CS_ACC[(size_t)bc * D_INNER + d];
#pragma unroll
    for (int n = 0; n < D_STATE; ++n) {
      const size_t idx = ((size_t)bc * D_STATE + n) * D_INNER + d;
      acc += h0[n] * CS_S[idx];
      h0[n] = CS_P[idx] * h0[n] + CS_H[idx];
    }
  }
  ybar[b * D_INNER + d] = acc * (1.f / (float)SEQL);
}

// ---------------------------------------------------------------------------
// K6: out[b,o] = ybar[b,:] @ Woc[:,o] + b_cls[o]    (32 x 6)
// ---------------------------------------------------------------------------
__global__ void k_final(const float* __restrict__ ybar, const float* __restrict__ Woc,
                        const float* __restrict__ b_cls, float* __restrict__ out) {
  int tid = threadIdx.x;
  if (tid >= BATCH * OUTC) return;
  int b = tid / OUTC, o = tid % OUTC;
  float acc = b_cls[o];
  for (int dd = 0; dd < D_INNER; ++dd) acc += ybar[b * D_INNER + dd] * Woc[dd * OUTC + o];
  out[tid] = acc;
}

// ---------------------------------------------------------------------------
extern "C" void kernel_launch(void* const* d_in, const int* in_sizes, int n_in,
                              void* d_out, int out_size, void* d_ws, size_t ws_size,
                              hipStream_t stream) {
  (void)in_sizes; (void)n_in; (void)out_size; (void)ws_size;
  const float* x        = (const float*)d_in[0];
  const float* W_in     = (const float*)d_in[1];
  const float* b_in     = (const float*)d_in[2];
  const float* W_inproj = (const float*)d_in[3];
  const float* conv_w   = (const float*)d_in[4];
  const float* conv_b   = (const float*)d_in[5];
  const float* W_xproj  = (const float*)d_in[6];
  const float* W_dt     = (const float*)d_in[7];
  const float* dt_bias  = (const float*)d_in[8];
  const float* A_log    = (const float*)d_in[9];
  const float* Dp       = (const float*)d_in[10];
  const float* W_out    = (const float*)d_in[11];
  const float* W_cls    = (const float*)d_in[12];
  const float* b_cls    = (const float*)d_in[13];

  char* ws = (char*)d_ws;
  size_t off = 0;
  auto take = [&](size_t bytes) -> char* {
    char* p = ws + off;
    off = (off + bytes + 255) & ~(size_t)255;
    return p;
  };
  _Float16* WfT  = (_Float16*)take((size_t)WFT_ELEMS * 2);
  float*    bfus = (float*)   take((size_t)XZ_COLS * 4);
  _Float16* WxT  = (_Float16*)take((size_t)WXT_ELEMS * 2);
  float*    Woc  = (float*)   take((size_t)D_INNER * OUTC * 4);
  _Float16* xh   = (_Float16*)take((size_t)BL * KPAD * 2);
  _Float16* xzh  = (_Float16*)take((size_t)BL * XZ_COLS * 2);
  _Float16* xch  = (_Float16*)take((size_t)BL * D_INNER * 2);
  float*    xdbl = (float*)   take((size_t)BL * XD_PAD * 4);
  const size_t cs_elems = (size_t)BATCH * NCHUNK * D_STATE * D_INNER;
  float*    CS_H = (float*)   take(cs_elems * 4);
  float*    CS_P = (float*)   take(cs_elems * 4);
  float*    CS_S = (float*)   take(cs_elems * 4);
  float*    CS_ACC = (float*) take((size_t)BATCH * NCHUNK * D_INNER * 4);
  float*    ybar = (float*)   take((size_t)BATCH * D_INNER * 4);

  const int prep_threads = WFT_ELEMS + XZ_COLS + WXT_ELEMS + D_INNER * OUTC;
  k_prep<<<(prep_threads + 255) / 256, 256, 0, stream>>>(W_in, b_in, W_inproj, W_xproj,
                                                         W_out, W_cls, WfT, bfus, WxT, Woc);
  k_padx<<<(BL * KPAD) / 256, 256, 0, stream>>>(x, xh);
  k_gemm_xz<<<dim3(XZ_COLS / 128, BL / 64), 256, 0, stream>>>(xh, WfT, bfus, xzh);
  k_conv<<<((size_t)BL * D_INNER) / 256, 256, 0, stream>>>(xzh, conv_w, conv_b, xch);
  k_gemm_xdbl<<<BL / 32, 256, 0, stream>>>(xch, WxT, xdbl);
  k_scan_chunk<<<dim3(2, NCHUNK, BATCH), 256, 0, stream>>>(xzh, xch, xdbl, W_dt, dt_bias,
                                                           A_log, Dp, CS_H, CS_P, CS_S, CS_ACC);
  k_scan_fix<<<BATCH, D_INNER, 0, stream>>>(CS_H, CS_P, CS_S, CS_ACC, ybar);
  k_final<<<1, 256, 0, stream>>>(ybar, Woc, b_cls, (float*)d_out);
}
